// CustomAttentionMarkov_5068061409376
// MI455X (gfx1250) — matrix operand order repaired and checked by reading, not compiled, not run on hardware
//
#include <hip/hip_runtime.h>

typedef __bf16 bf16_t;
typedef __attribute__((ext_vector_type(16))) __bf16 v16bf;
typedef __attribute__((ext_vector_type(8)))  __bf16 v8bf;
typedef __attribute__((ext_vector_type(4)))  __bf16 v4bf;
typedef __attribute__((ext_vector_type(8)))  float  v8f;
typedef __attribute__((ext_vector_type(4)))  unsigned int u32x4;
typedef __attribute__((ext_vector_type(8)))  int i32x8;
typedef __attribute__((ext_vector_type(4)))  int i32x4;

static __device__ __forceinline__ bf16_t f2bf(float x) {
  unsigned int u = __builtin_bit_cast(unsigned int, x);
  unsigned int r = (u + 0x7FFFu + ((u >> 16) & 1u)) >> 16;
  return __builtin_bit_cast(bf16_t, (unsigned short)r);
}

static __device__ __forceinline__ v16bf cat16(v8bf lo, v8bf hi) {
  return __builtin_shufflevector(lo, hi, 0,1,2,3,4,5,6,7,8,9,10,11,12,13,14,15);
}

// A-fragment (16x32 bf16, MxK): lane covers row = lane&15.
// element j  <->  K = (j>>3)*16 + half*8 + (j&7)
static __device__ __forceinline__ v16bf load_frag_a(const bf16_t* rowp, int half) {
  v8bf lo = *(const v8bf*)(rowp + (half << 3));
  v8bf hi = *(const v8bf*)(rowp + 16 + (half << 3));
  return cat16(lo, hi);
}

// B-fragment (32x16 bf16, KxN) from B^T rows: lane covers col N = lane&15.
// element j  <->  K = (j>>3)*16 + half*8 + (j&7)   (B^T row contiguous in K)
static __device__ __forceinline__ v16bf load_frag_b(const bf16_t* btrowp, int half) {
  v8bf lo = *(const v8bf*)(btrowp + (half << 3));
  v8bf hi = *(const v8bf*)(btrowp + 16 + (half << 3));
  return cat16(lo, hi);
}

static __device__ __forceinline__ v8f wmma_bf16(v16bf a, v16bf b, v8f c) {
  return __builtin_amdgcn_wmma_f32_16x16x32_bf16(false, a, false, b, (short)0, c,
                                                 false, false);
}

// ---------------------------------------------------------------------------
// TDM: DMA one 128x32 bf16 tile of A[8192][1024] into LDS with padded rows.
// D# per CDNA5 ISA:
//  group0: {count=1 | type=2, lds_addr, global_addr lo, global_addr hi}
//  group1: data_size=2B, pad_enable, pad_interval=64B (code 3),
//          pad_amount=4 dwords (code 3)  ->  LDS row stride 80 B == LDA=40,
//          tensor_dim0=1024, tensor_dim1=8192, tile_dim0=32, tile_dim1=128,
//          tensor_dim0_stride=1024, workgroup_mask=0 (not in cluster).
// Issued by one wave; completion via s_wait_tensorcnt 0 then the barrier.
// ---------------------------------------------------------------------------
static __device__ __forceinline__ void tdm_load_tile_a(const bf16_t* gsrc,
                                                       unsigned lds_off) {
#if __has_builtin(__builtin_amdgcn_tensor_load_to_lds)
  unsigned long long ga = (unsigned long long)(uintptr_t)gsrc;
  u32x4 g0 = { 1u,                                   // count=1, user mode
               lds_off,                              // lds_addr [63:32]
               (unsigned)ga,                         // global_addr [95:64]
               (unsigned)((ga >> 32) & 0x01FFFFFFull) | (2u << 30) }; // type=2
  i32x8 g1 = { (int)((1u << 16) | (1u << 20) | (3u << 22) | (3u << 25)),
               (int)(1024u << 16),                   // tensor_dim0[15:0]
               (int)(8192u << 16),                   // tensor_dim1[15:0]
               (int)(32u << 16),                     // tile_dim0
               (int)128,                             // tile_dim1
               (int)1024,                            // tensor_dim0_stride
               0, 0 };
  i32x4 z4 = {};
#if __has_include(<hip/amd_detail/amd_gfx1250_TDM.h>)
  i32x8 z8 = {};
  __builtin_amdgcn_tensor_load_to_lds(g0, g1, z4, z4, z8, 0);
#else
  __builtin_amdgcn_tensor_load_to_lds(g0, g1, z4, z4, 0);
#endif
  __builtin_amdgcn_s_wait_tensorcnt(0);
#endif
}

// ---------------------------------------------------------------------------
// GEMM: C[M=8192][1024] = A[8192][1024] * W[1024][1024] + bias
// IN_BF16: 0 = A is f32 (convert to bf16 in LDS), 1 = A is bf16 (TDM-staged)
// OUT_MODE: 0 = bf16 row-major, 1 = bf16 transposed per batch (vt[b][n][s]),
//           2 = f32 row-major
// Block tile 128x128x32, 256 threads = 8 waves in a 4(M) x 2(N) grid,
// each wave computes 32x64 via 2x4 v_wmma_f32_16x16x32_bf16 accumulators.
// ---------------------------------------------------------------------------
template<int IN_BF16, int OUT_MODE>
__global__ __launch_bounds__(256)
void gemm_bf16_kernel(const void* __restrict__ Aptr,
                      const float* __restrict__ W,
                      const float* __restrict__ bias,
                      void* __restrict__ Cptr) {
  constexpr int BK = 32;
  constexpr int LDA = 40;   // padded bf16 elems per row (80 B, 16B aligned)
  constexpr int LDB = 40;
  __shared__ bf16_t As[128 * LDA];   // X tile, row-major [m][k]
  __shared__ bf16_t Bs[128 * LDB];   // W tile transposed: Bs[n][k] = W[k][n]

  const int tid  = threadIdx.x;
  const int lane = tid & 31;
  const int wid  = tid >> 5;
  const int half = lane >> 4;
  const int l15  = lane & 15;
  const int waveM = wid >> 1;
  const int waveN = wid & 1;
  const int blockN = blockIdx.x * 128;
  const int blockM = blockIdx.y * 128;

  v8f acc[2][4];
#pragma unroll
  for (int m = 0; m < 2; ++m)
#pragma unroll
    for (int n = 0; n < 4; ++n) acc[m][n] = {};

  const float*  Af = (const float*)Aptr;
  const bf16_t* Ab = (const bf16_t*)Aptr;

  for (int k0 = 0; k0 < 1024; k0 += BK) {
    // ---- load X tile (128 x 32) into LDS as bf16 ----
    if (IN_BF16) {
#if __has_builtin(__builtin_amdgcn_tensor_load_to_lds)
      if (wid == 0)
        tdm_load_tile_a(Ab + (size_t)blockM * 1024 + k0,
                        (unsigned)(uintptr_t)&As[0]);
#else
#pragma unroll
      for (int i = 0; i < 2; ++i) {
        int c = tid + i * 256;           // 512 chunks of 8 bf16
        int row = c >> 2, c8 = c & 3;
        v8bf v = *(const v8bf*)(Ab + (size_t)(blockM + row) * 1024 + k0 + c8 * 8);
        *(v8bf*)&As[row * LDA + c8 * 8] = v;
      }
#endif
    } else {
#pragma unroll
      for (int i = 0; i < 4; ++i) {
        int f = tid + i * 256;           // 1024 float4s
        int row = f >> 3, c4 = f & 7;
        float4 v = ((const float4*)(Af + (size_t)(blockM + row) * 1024 + k0))[c4];
        v4bf t; t[0] = f2bf(v.x); t[1] = f2bf(v.y); t[2] = f2bf(v.z); t[3] = f2bf(v.w);
        *(v4bf*)&As[row * LDA + c4 * 4] = t;
      }
    }
    // ---- load W tile (32 x 128) transposed into LDS ----
#pragma unroll
    for (int i = 0; i < 4; ++i) {
      int f = tid + i * 256;             // 1024 float4s over 32x128
      int kr = f >> 5, c4 = f & 31;
      float4 v = ((const float4*)(W + (size_t)(k0 + kr) * 1024 + blockN))[c4];
      Bs[(c4 * 4 + 0) * LDB + kr] = f2bf(v.x);
      Bs[(c4 * 4 + 1) * LDB + kr] = f2bf(v.y);
      Bs[(c4 * 4 + 2) * LDB + kr] = f2bf(v.z);
      Bs[(c4 * 4 + 3) * LDB + kr] = f2bf(v.w);
    }
    __syncthreads();

    v16bf aF[2], bF[4];
#pragma unroll
    for (int m = 0; m < 2; ++m)
      aF[m] = load_frag_a(&As[(waveM * 32 + m * 16 + l15) * LDA], half);
#pragma unroll
    for (int n = 0; n < 4; ++n)
      bF[n] = load_frag_b(&Bs[(waveN * 64 + n * 16 + l15) * LDB], half);
#pragma unroll
    for (int m = 0; m < 2; ++m)
#pragma unroll
      for (int n = 0; n < 4; ++n)
        acc[m][n] = wmma_bf16(aF[m], bF[n], acc[m][n]);
    __syncthreads();
  }

  // ---- epilogue: bias + store ----
#pragma unroll
  for (int m = 0; m < 2; ++m) {
#pragma unroll
    for (int n = 0; n < 4; ++n) {
      int col = blockN + waveN * 64 + n * 16 + l15;
      float bv = bias[col];
#pragma unroll
      for (int r = 0; r < 8; ++r) {
        int row = blockM + waveM * 32 + m * 16 + half * 8 + r;
        float v = acc[m][n][r] + bv;
        if (OUT_MODE == 0) {
          ((bf16_t*)Cptr)[(size_t)row * 1024 + col] = f2bf(v);
        } else if (OUT_MODE == 1) {
          int b = row >> 11, s = row & 2047;          // S = 2048
          ((bf16_t*)Cptr)[((size_t)b * 1024 + col) * 2048 + s] = f2bf(v);
        } else {
          ((float*)Cptr)[(size_t)row * 1024 + col] = v;
        }
      }
    }
  }
}

// ---------------------------------------------------------------------------
// Flash attention: per workgroup = 16 query rows of one batch, 8 waves.
// Key tiles of 128: wave w owns score frag cols [w*16, w*16+16) and output
// d-columns [w*128, (w+1)*128) as 8 f32 WMMA accumulators.
// q,k: bf16 [B][S][D] row-major;  vt: bf16 [B][D][S] (V transposed).
// ---------------------------------------------------------------------------
__global__ __launch_bounds__(256)
void attn_kernel(const bf16_t* __restrict__ q, const bf16_t* __restrict__ k,
                 const bf16_t* __restrict__ vt, bf16_t* __restrict__ outp) {
  constexpr int S = 2048, D = 1024;
  constexpr int PST = 136;                 // padded P row stride (272 B)
  __shared__ float redmax[8][16];
  __shared__ float redsum[8][16];
  __shared__ float m_run[16], l_run[16], scale_s[16], mnew_s[16];
  __shared__ bf16_t P[16 * PST];

  const int tid  = threadIdx.x;
  const int lane = tid & 31;
  const int w    = tid >> 5;
  const int half = lane >> 4;
  const int l15  = lane & 15;
  const int b     = blockIdx.y;
  const int qBase = blockIdx.x * 16;

  if (tid < 16) { m_run[tid] = -1e30f; l_run[tid] = 0.f; }
  __syncthreads();

  v8f o[8];
#pragma unroll
  for (int nt = 0; nt < 8; ++nt) o[nt] = {};

  const bf16_t* qrow = q + ((size_t)b * S + qBase + l15) * D;
  const int nkt = (qBase >> 7) + 1;
  const float rscale = 0.03125f;           // 1/sqrt(1024)

  for (int kt = 0; kt < nkt; ++kt) {
    const int ktBase = kt << 7;
    const bf16_t* krow = k + ((size_t)b * S + ktBase + w * 16 + l15) * D;

    // hint next key tile's rows into cache (global_prefetch_b8)
    if (kt + 1 < nkt) __builtin_prefetch(krow + (size_t)128 * D, 0, 1);

    // ---- scores: S(16x16 per wave) = Q(16xD) . K^T, K rows are B^T rows ----
    v8f s = {};
#pragma unroll 8
    for (int d = 0; d < D; d += 32) {
      v16bf aF = load_frag_a(qrow + d, half);
      v16bf bF = load_frag_b(krow + d, half);
      s = wmma_bf16(aF, bF, s);
    }

    // ---- scale + causal mask + per-row max (rows = half*8+r over 16 lanes) --
    const int col = ktBase + w * 16 + l15;
#pragma unroll
    for (int r = 0; r < 8; ++r) {
      int row = half * 8 + r;
      float val = s[r] * rscale;
      if (col > qBase + row) val = -1e9f;
      s[r] = val;
      float mx = val;
      mx = fmaxf(mx, __shfl_xor(mx, 1, 32));
      mx = fmaxf(mx, __shfl_xor(mx, 2, 32));
      mx = fmaxf(mx, __shfl_xor(mx, 4, 32));
      mx = fmaxf(mx, __shfl_xor(mx, 8, 32));
      if (l15 == 0) redmax[w][row] = mx;
    }
    __syncthreads();

    if (tid < 16) {
      float tm = redmax[0][tid];
#pragma unroll
      for (int ww = 1; ww < 8; ++ww) tm = fmaxf(tm, redmax[ww][tid]);
      float mo = m_run[tid];
      float mn = fmaxf(mo, tm);
      m_run[tid]   = mn;
      mnew_s[tid]  = mn;
      scale_s[tid] = __expf(mo - mn);      // first tile: exp(-inf)=0
    }
    __syncthreads();

    // ---- P = exp(S-m), row sums, rescale O, stage P (bf16) in LDS ----
#pragma unroll
    for (int r = 0; r < 8; ++r) {
      int row = half * 8 + r;
      float p = __expf(s[r] - mnew_s[row]);
      P[row * PST + w * 16 + l15] = f2bf(p);
      float sum = p;
      sum += __shfl_xor(sum, 1, 32);
      sum += __shfl_xor(sum, 2, 32);
      sum += __shfl_xor(sum, 4, 32);
      sum += __shfl_xor(sum, 8, 32);
      if (l15 == 0) redsum[w][row] = sum;
      float sc = scale_s[row];
#pragma unroll
      for (int nt = 0; nt < 8; ++nt) o[nt][r] *= sc;
    }
    __syncthreads();

    if (tid < 16) {
      float ts = redsum[0][tid];
#pragma unroll
      for (int ww = 1; ww < 8; ++ww) ts += redsum[ww][tid];
      l_run[tid] = l_run[tid] * scale_s[tid] + ts;
    }

    // ---- O += P(16x128) @ V(128xD): B-frags straight from vt (contiguous) --
#pragma unroll
    for (int kk = 0; kk < 4; ++kk) {
      v16bf aF = load_frag_a(&P[l15 * PST + kk * 32], half);
#pragma unroll
      for (int nt = 0; nt < 8; ++nt) {
        int n = w * 128 + nt * 16 + l15;
        const bf16_t* vrow = vt + ((size_t)b * D + n) * S + ktBase + kk * 32;
        v16bf bF = load_frag_b(vrow, half);
        o[nt] = wmma_bf16(aF, bF, o[nt]);
      }
    }
    __syncthreads();
  }

  // ---- finalize: O / l, store bf16 ----
#pragma unroll
  for (int r = 0; r < 8; ++r) {
    int row = half * 8 + r;
    float inv = 1.0f / l_run[row];
    size_t rbase = ((size_t)b * S + qBase + row) * D;
#pragma unroll
    for (int nt = 0; nt < 8; ++nt) {
      int n = w * 128 + nt * 16 + l15;
      outp[rbase + n] = f2bf(o[nt][r] * inv);
    }
  }
}

// ---------------------------------------------------------------------------
extern "C" void kernel_launch(void* const* d_in, const int* in_sizes, int n_in,
                              void* d_out, int out_size, void* d_ws, size_t ws_size,
                              hipStream_t stream) {
  const float* query = (const float*)d_in[0];
  const float* key_t = (const float*)d_in[1];
  const float* value = (const float*)d_in[2];
  const float* Wq = (const float*)d_in[3];
  const float* bq = (const float*)d_in[4];
  const float* Wk = (const float*)d_in[5];
  const float* bk = (const float*)d_in[6];
  const float* Wv = (const float*)d_in[7];
  const float* bv = (const float*)d_in[8];
  const float* Wo = (const float*)d_in[9];
  const float* bo = (const float*)d_in[10];

  const size_t elems = (size_t)4 * 2048 * 1024;   // B*S*D
  bf16_t* qw  = (bf16_t*)d_ws;
  bf16_t* kw  = qw  + elems;
  bf16_t* vtw = kw  + elems;       // V transposed: [B][D][S]
  bf16_t* aw  = vtw + elems;       // attention output, bf16 [B][S][D]

  dim3 gg(8, 64), bb(256);
  gemm_bf16_kernel<0, 0><<<gg, bb, 0, stream>>>(query, Wq, bq, qw);
  gemm_bf16_kernel<0, 0><<<gg, bb, 0, stream>>>(key_t, Wk, bk, kw);
  gemm_bf16_kernel<0, 1><<<gg, bb, 0, stream>>>(value, Wv, bv, vtw);
  attn_kernel<<<dim3(128, 4), bb, 0, stream>>>(qw, kw, vtw, aw);
  gemm_bf16_kernel<1, 2><<<gg, bb, 0, stream>>>(aw, Wo, bo, (float*)d_out);
}
